// SeqDecoder_21217138442907
// MI455X (gfx1250) — matrix operand order repaired and checked by reading, not compiled, not run on hardware
//
#include <hip/hip_runtime.h>
#include <hip/hip_bf16.h>

// Problem constants (from reference)
#define N_   16
#define L_   64
#define LP_  256
#define D_   512
#define OPS_ 32
#define NUM_ 64
#define V_   96
#define G3_  (3 * D_)          // 1536
#define NEG_INF (-1.0e12f)
#define LOGITS_ELEMS (N_ * L_ * V_)   // 98304

typedef __attribute__((ext_vector_type(16))) __bf16 v16bf;
typedef __attribute__((ext_vector_type(8)))  float  v8f;

// ---------------- helpers ----------------

static __device__ __forceinline__ __bf16 f2bf(float f) {
  unsigned u = __builtin_bit_cast(unsigned, f);
  u += 0x7FFFu + ((u >> 16) & 1u);           // round-to-nearest-even
  unsigned short h = (unsigned short)(u >> 16);
  return __builtin_bit_cast(__bf16, h);
}

// CDNA5 has a hardware V_TANH_F32 transcendental; use it when clang exposes it.
#if defined(__has_builtin)
#if __has_builtin(__builtin_amdgcn_tanhf)
#define HAS_HW_TANH 1
#endif
#endif

static __device__ __forceinline__ float fast_tanh(float x) {
#ifdef HAS_HW_TANH
  return __builtin_amdgcn_tanhf(x);
#else
  return tanhf(x);
#endif
}

static __device__ __forceinline__ float fast_sigmoid(float x) {
  // sigmoid(x) = 0.5 + 0.5*tanh(x/2) -> avoids the IEEE divide expansion
  return 0.5f + 0.5f * fast_tanh(0.5f * x);
}

static __device__ __forceinline__ v8f vzero8() {
  v8f z;
#pragma unroll
  for (int i = 0; i < 8; ++i) z[i] = 0.0f;
  return z;
}

// A fragment: 16x32 bf16, rows M=0..15.  Contiguous K order used here (B frags match):
// lanes 0-15: row=lane,    K = 0..15
// lanes 16-31: row=lane-16, K = 16..31
static __device__ __forceinline__ v16bf load_a_frag(const __bf16* A, int lda,
                                                    int row0, int k0) {
  int lane = threadIdx.x & 31;
  const __bf16* p = A + (size_t)(row0 + (lane & 15)) * lda + k0 + ((lane >> 4) << 4);
  v16bf f;
#pragma unroll
  for (int i = 0; i < 8; ++i) f[i] = p[i];
#pragma unroll
  for (int i = 0; i < 8; ++i) f[8 + i] = p[8 + i];
  return f;
}

// B fragment for A @ W^T:  B[k][n] = W[n][k], W row-major [Nout, K].
// lanes 0-15: col=lane, K=0..15 ; lanes 16-31: col=lane-16, K=16..31
static __device__ __forceinline__ v16bf load_bt_frag(const __bf16* W, int ldw,
                                                     int n0, int k0) {
  int lane = threadIdx.x & 31;
  const __bf16* p = W + (size_t)(n0 + (lane & 15)) * ldw + k0 + ((lane >> 4) << 4);
  v16bf f;
#pragma unroll
  for (int i = 0; i < 16; ++i) f[i] = p[i];
  return f;
}

// B fragment from a row-major B[K][N] matrix directly.
static __device__ __forceinline__ v16bf load_b_frag(const __bf16* B, int ldb,
                                                    int k0, int n0) {
  int lane = threadIdx.x & 31;
  int n  = n0 + (lane & 15);
  int kb = k0 + ((lane >> 4) << 4);
  v16bf f;
#pragma unroll
  for (int i = 0; i < 16; ++i) f[i] = B[(size_t)(kb + i) * ldb + n];
  return f;
}

static __device__ __forceinline__ v8f wmma_bf16(v16bf a, v16bf b, v8f c) {
  return __builtin_amdgcn_wmma_f32_16x16x32_bf16(false, a, false, b,
                                                 (short)0, c, false, false);
}

static __device__ __forceinline__ void wait_asynccnt0() {
#if defined(__has_builtin) && __has_builtin(__builtin_amdgcn_s_wait_asynccnt)
  __builtin_amdgcn_s_wait_asynccnt(0);
#else
  asm volatile("s_wait_asynccnt 0" ::: "memory");
#endif
}

// ---------------- kernels ----------------

__global__ void cvt_bf16_kernel(const float* __restrict__ src,
                                __bf16* __restrict__ dst, int n) {
  int i = blockIdx.x * blockDim.x + threadIdx.x;
  if (i < n) dst[i] = f2bf(src[i]);
}

// Embedding gather: dec_bf[n*L+l, :] = (id<OPS ? W_ops0[id] : W_nums[n, id-OPS])
__global__ void embed_kernel(const int* __restrict__ ids,
                             const float* __restrict__ Wops0,
                             const float* __restrict__ Wnums,
                             __bf16* __restrict__ dec_bf) {
  int row = blockIdx.x;             // n*L + l
  int n   = row / L_;
  int id  = ids[row];
  const float* src = (id < OPS_) ? (Wops0 + (size_t)id * D_)
                                 : (Wnums + (size_t)(n * NUM_ + (id - OPS_)) * D_);
  for (int i = threadIdx.x; i < D_; i += blockDim.x)
    dec_bf[(size_t)row * D_ + i] = f2bf(src[i]);
}

// Generic GEMM: C = act(A @ W^T + bias). A [M,K] bf16, W [Nout,K] bf16.
// Block tile 128x64, 8 waves as 4x2, each wave 32x32 (2x2 WMMA tiles).
template <int ACT>  // 0 = none, 1 = tanh
__global__ __launch_bounds__(256) void gemm_awt_kernel(
    const __bf16* __restrict__ A, int lda,
    const __bf16* __restrict__ W, int ldw,
    const float* __restrict__ bias,
    float* __restrict__ C, int ldc,
    __bf16* __restrict__ Cbf, int ldcb,
    int K) {
  int wave = threadIdx.x >> 5;
  int wr = wave >> 1, wc = wave & 1;
  int m0 = blockIdx.x * 128 + wr * 32;
  int n0 = blockIdx.y * 64 + wc * 32;

  v8f acc[2][2];
#pragma unroll
  for (int i = 0; i < 2; ++i)
#pragma unroll
    for (int j = 0; j < 2; ++j) acc[i][j] = vzero8();

  for (int k0 = 0; k0 < K; k0 += 32) {
    v16bf a0 = load_a_frag(A, lda, m0, k0);
    v16bf a1 = load_a_frag(A, lda, m0 + 16, k0);
    v16bf b0 = load_bt_frag(W, ldw, n0, k0);
    v16bf b1 = load_bt_frag(W, ldw, n0 + 16, k0);
    acc[0][0] = wmma_bf16(a0, b0, acc[0][0]);
    acc[0][1] = wmma_bf16(a0, b1, acc[0][1]);
    acc[1][0] = wmma_bf16(a1, b0, acc[1][0]);
    acc[1][1] = wmma_bf16(a1, b1, acc[1][1]);
  }

  int lane  = threadIdx.x & 31;
  int colin = lane & 15;
  int rbase = (lane >> 4) << 3;
#pragma unroll
  for (int i = 0; i < 2; ++i)
#pragma unroll
    for (int j = 0; j < 2; ++j) {
      int col = n0 + j * 16 + colin;
      float bv = bias ? bias[col] : 0.0f;
#pragma unroll
      for (int v = 0; v < 8; ++v) {
        int row = m0 + i * 16 + rbase + v;
        float x = acc[i][j][v] + bv;
        if (ACT == 1) x = fast_tanh(x);
        if (C)   C[(size_t)row * ldc + col] = x;
        if (Cbf) Cbf[(size_t)row * ldcb + col] = f2bf(x);
      }
    }
}

// GRU scan: one workgroup (32 waves), h [16,512] resident in LDS.
// Each wave owns a 16-wide column slice and computes 3 gate tiles (r,z,n)
// per timestep via 16x16x32 bf16 WMMA over K=512.
__global__ __launch_bounds__(1024) void gru_kernel(
    const float* __restrict__ gi,     // [N*L, 1536], row = n*L + t (bih folded in)
    const float* __restrict__ h0,     // past_value [N, D]
    const __bf16* __restrict__ Whh,   // [1536, 512] bf16
    const float* __restrict__ bhh,    // [1536]
    __bf16* __restrict__ qscs,        // [N*L, 1024]; qs -> cols [0,512)
    float* __restrict__ hn_out) {     // [N, D]
  __shared__ __bf16 hbf[N_][D_];
  __shared__ float  hf[N_][D_];

  int tid = threadIdx.x;
  for (int i = tid; i < N_ * D_; i += blockDim.x) {
    float v = h0[i];
    hf[i >> 9][i & (D_ - 1)]  = v;
    hbf[i >> 9][i & (D_ - 1)] = f2bf(v);
  }
  __syncthreads();

  int wave  = tid >> 5;
  int lane  = tid & 31;
  int c     = wave * 16;           // column slice base (0..496)
  int colin = lane & 15;
  int rbase = (lane >> 4) << 3;

  float bhr = bhh[c + colin];
  float bhz = bhh[D_ + c + colin];
  float bhn = bhh[2 * D_ + c + colin];

  for (int t = 0; t < L_; ++t) {
    v8f accr = vzero8(), accz = vzero8(), accn = vzero8();
    for (int k0 = 0; k0 < D_; k0 += 32) {
      v16bf a  = load_a_frag(&hbf[0][0], D_, 0, k0);
      v16bf br = load_bt_frag(Whh, D_, c, k0);
      v16bf bz = load_bt_frag(Whh, D_, D_ + c, k0);
      v16bf bn = load_bt_frag(Whh, D_, 2 * D_ + c, k0);
      accr = wmma_bf16(a, br, accr);
      accz = wmma_bf16(a, bz, accz);
      accn = wmma_bf16(a, bn, accn);
    }

    float hnew[8];
#pragma unroll
    for (int v = 0; v < 8; ++v) {
      int m = rbase + v;                       // batch index 0..15
      const float* gr = gi + (size_t)(m * L_ + t) * G3_;
      float r  = fast_sigmoid(gr[c + colin] + accr[v] + bhr);
      float z  = fast_sigmoid(gr[D_ + c + colin] + accz[v] + bhz);
      float nn = fast_tanh(gr[2 * D_ + c + colin] + r * (accn[v] + bhn));
      hnew[v] = (1.0f - z) * nn + z * hf[m][c + colin];
    }
    __syncthreads();   // all reads of h done
#pragma unroll
    for (int v = 0; v < 8; ++v) {
      int m = rbase + v;
      float x = hnew[v];
      hf[m][c + colin]  = x;
      __bf16 xb = f2bf(x);
      hbf[m][c + colin] = xb;
      qscs[(size_t)(m * L_ + t) * 1024 + c + colin] = xb;   // qs output
    }
    __syncthreads();   // writes visible before next step
  }

  for (int i = tid; i < N_ * D_; i += blockDim.x)
    hn_out[i] = hf[i >> 9][i & (D_ - 1)];
}

// scores[n,q,p] = attn_V . tanh(sx[n,q,:] + sm[n,p,:]) + attn_Vb  (masked)
// Block = 8 waves handling 8 consecutive q; sm rows staged into LDS in tiles
// of 8 via CDNA5 GLOBAL_LOAD_ASYNC_TO_LDS_B128 (ASYNCcnt).
#define QTILE_ 8
#define PTILE_ 8
__global__ __launch_bounds__(256) void scores_kernel(
    const float* __restrict__ sx,    // [N*L, D]
    const float* __restrict__ sm,    // [N*LP, D]
    const float* __restrict__ aV,    // [D]
    const float* __restrict__ aVb,   // [1]
    const unsigned char* __restrict__ amask, // [N, LP]
    float* __restrict__ scores) {    // [N*L, LP]
  __shared__ float smtile[PTILE_][D_];   // 16 KB; only LDS alloc -> offset 0

  int block = blockIdx.x;                // N * (L/QTILE) blocks
  int n  = block / (L_ / QTILE_);
  int q0 = (block % (L_ / QTILE_)) * QTILE_;
  int wave = threadIdx.x >> 5;
  int lane = threadIdx.x & 31;
  int q = q0 + wave;                     // one q per wave

  // cache this wave's sx row and attn_V in registers (16 elems per lane)
  float sxr[16], aVr[16];
  const float* sxrow = sx + (size_t)(n * L_ + q) * D_;
#pragma unroll
  for (int j = 0; j < 16; ++j) {
    sxr[j] = sxrow[lane + 32 * j];
    aVr[j] = aV[lane + 32 * j];
  }
  float vb = aVb[0];

  for (int p0 = 0; p0 < LP_; p0 += PTILE_) {
    // async-stage 8 contiguous sm rows (16 KB) into LDS: 1024 b128 transfers
    const float* gbase = sm + (size_t)(n * LP_ + p0) * D_;
#pragma unroll
    for (int it = 0; it < 4; ++it) {
      int idx = (int)threadIdx.x + 256 * it;      // b128 transfer index
      unsigned ldsoff = (unsigned)(idx * 16);
      const float* gp = gbase + idx * 4;
      asm volatile("global_load_async_to_lds_b128 %0, %1, off"
                   :: "v"(ldsoff), "v"(gp) : "memory");
    }
    wait_asynccnt0();
    __syncthreads();                               // tile visible to all waves

#pragma unroll
    for (int pp = 0; pp < PTILE_; ++pp) {
      float s = 0.0f;
#pragma unroll
      for (int j = 0; j < 16; ++j)
        s += aVr[j] * fast_tanh(sxr[j] + smtile[pp][lane + 32 * j]);
#pragma unroll
      for (int off = 16; off > 0; off >>= 1)
        s += __shfl_xor(s, off, 32);
      if (lane == 0) {
        int p = p0 + pp;
        float val = s + vb;
        if (!amask[n * LP_ + p]) val = NEG_INF;
        scores[(size_t)(n * L_ + q) * LP_ + p] = val;
      }
    }
    __syncthreads();                               // reads done before re-stage
  }
}

// Row softmax over LP=256 (blockDim.x == 256), emit bf16 probabilities.
__global__ __launch_bounds__(256) void softmax_kernel(
    const float* __restrict__ scores, __bf16* __restrict__ wbf) {
  int row = blockIdx.x;
  int p   = threadIdx.x;
  float v = scores[(size_t)row * LP_ + p];
  __shared__ float red[256];
  red[p] = v; __syncthreads();
  for (int s = 128; s > 0; s >>= 1) {
    if (p < s) red[p] = fmaxf(red[p], red[p + s]);
    __syncthreads();
  }
  float mx = red[0]; __syncthreads();
  float e = __expf(v - mx);
  red[p] = e; __syncthreads();
  for (int s = 128; s > 0; s >>= 1) {
    if (p < s) red[p] += red[p + s];
    __syncthreads();
  }
  float inv = 1.0f / red[0];
  wbf[(size_t)row * LP_ + p] = f2bf(e * inv);
}

// cs[n] = w[n] @ enc[n] : [64,256] x [256,512], written into qscs cols [512,1024)
__global__ __launch_bounds__(256) void cs_kernel(
    const __bf16* __restrict__ wsoft,  // [N*L, LP]
    const __bf16* __restrict__ encbf,  // [N*LP, D]
    __bf16* __restrict__ qscs) {       // [N*L, 1024]
  int n    = blockIdx.y;
  int wave = threadIdx.x >> 5;
  int wr = wave >> 1, wc = wave & 1;
  int m0 = wr * 16;
  int n0 = blockIdx.x * 64 + wc * 32;
  const __bf16* A = wsoft + (size_t)n * L_ * LP_;
  const __bf16* B = encbf + (size_t)n * LP_ * D_;

  v8f acc[2] = {vzero8(), vzero8()};
  for (int k0 = 0; k0 < LP_; k0 += 32) {
    v16bf a  = load_a_frag(A, LP_, m0, k0);
    v16bf b0 = load_b_frag(B, D_, k0, n0);
    v16bf b1 = load_b_frag(B, D_, k0, n0 + 16);
    acc[0] = wmma_bf16(a, b0, acc[0]);
    acc[1] = wmma_bf16(a, b1, acc[1]);
  }

  int lane = threadIdx.x & 31;
  int colin = lane & 15, rbase = (lane >> 4) << 3;
#pragma unroll
  for (int j = 0; j < 2; ++j)
#pragma unroll
    for (int v = 0; v < 8; ++v) {
      int row = n * L_ + m0 + rbase + v;
      int col = D_ + n0 + j * 16 + colin;
      qscs[(size_t)row * 1024 + col] = f2bf(acc[j][v]);
    }
}

// logits[n] = feat[n] @ wn[n]^T, wn rows: j<32 -> W_ops1[j], else W_nums[n,j-32]
// Block per n: 8 waves as 4(M) x 2(V), each wave 16x48 (3 tiles). Masked fp32 out.
__global__ __launch_bounds__(256) void logits_kernel(
    const __bf16* __restrict__ featbf,   // [N*L, D]
    const __bf16* __restrict__ Wops1bf,  // [OPS, D]
    const __bf16* __restrict__ Wnumsbf,  // [N*NUM, D]
    const unsigned char* __restrict__ omask, // [N, V]
    float* __restrict__ out) {           // [N, L, V]
  int n    = blockIdx.x;
  int wave = threadIdx.x >> 5;
  int wr = wave >> 1, wc = wave & 1;
  int m0 = wr * 16;
  int v0 = wc * 48;
  const __bf16* A = featbf + (size_t)n * L_ * D_;
  int lane = threadIdx.x & 31;

  v8f acc[3] = {vzero8(), vzero8(), vzero8()};
  for (int k0 = 0; k0 < D_; k0 += 32) {
    v16bf a = load_a_frag(A, D_, m0, k0);
#pragma unroll
    for (int j = 0; j < 3; ++j) {
      int jrow = v0 + j * 16 + (lane & 15);
      const __bf16* rp = (jrow < OPS_)
          ? (Wops1bf + (size_t)jrow * D_)
          : (Wnumsbf + (size_t)(n * NUM_ + jrow - OPS_) * D_);
      const __bf16* p = rp + k0 + ((lane >> 4) << 4);
      v16bf b;
#pragma unroll
      for (int i = 0; i < 16; ++i) b[i] = p[i];
      acc[j] = wmma_bf16(a, b, acc[j]);
    }
  }

  int colin = lane & 15, rbase = (lane >> 4) << 3;
#pragma unroll
  for (int j = 0; j < 3; ++j) {
    int vv = v0 + j * 16 + colin;
    bool keep = omask[n * V_ + vv] != 0;
#pragma unroll
    for (int v = 0; v < 8; ++v) {
      int q = m0 + rbase + v;
      float x = keep ? acc[j][v] : NEG_INF;
      out[(size_t)(n * L_ + q) * V_ + vv] = x;
    }
  }
}

// ---------------- launcher ----------------

extern "C" void kernel_launch(void* const* d_in, const int* in_sizes, int n_in,
                              void* d_out, int out_size, void* d_ws, size_t ws_size,
                              hipStream_t stream) {
  const int*   ids    = (const int*)d_in[0];
  const float* h0     = (const float*)d_in[1];
  const float* Wnums  = (const float*)d_in[2];
  const unsigned char* omask = (const unsigned char*)d_in[3];
  const float* enc    = (const float*)d_in[4];
  const unsigned char* amask = (const unsigned char*)d_in[5];
  const float* Wops0  = (const float*)d_in[6];
  const float* Wops1  = (const float*)d_in[7];
  const float* attnW  = (const float*)d_in[8];
  const float* attnWb = (const float*)d_in[9];
  const float* attnV  = (const float*)d_in[10];
  const float* attnVb = (const float*)d_in[11];
  const float* Wih    = (const float*)d_in[12];
  const float* Whh    = (const float*)d_in[13];
  const float* bih    = (const float*)d_in[14];
  const float* bhh    = (const float*)d_in[15];
  const float* fcW    = (const float*)d_in[16];
  const float* fcb    = (const float*)d_in[17];

  // workspace bump allocator (256B aligned)
  char* p = (char*)d_ws;
  auto alloc = [&](size_t bytes) -> char* {
    char* r = p;
    p += (bytes + 255) & ~(size_t)255;
    return r;
  };
  float*  gi       = (float*) alloc((size_t)N_ * L_ * G3_ * 4);    // 6 MB
  float*  sx       = (float*) alloc((size_t)N_ * L_ * D_ * 4);     // 2 MB
  float*  sm       = (float*) alloc((size_t)N_ * LP_ * D_ * 4);    // 8 MB
  float*  scores   = (float*) alloc((size_t)N_ * L_ * LP_ * 4);    // 1 MB
  __bf16* wih_bf   = (__bf16*)alloc((size_t)G3_ * D_ * 2);
  __bf16* whh_bf   = (__bf16*)alloc((size_t)G3_ * D_ * 2);
  __bf16* attnw_bf = (__bf16*)alloc((size_t)D_ * 2 * D_ * 2);
  __bf16* fcw_bf   = (__bf16*)alloc((size_t)D_ * 2 * D_ * 2);
  __bf16* enc_bf   = (__bf16*)alloc((size_t)N_ * LP_ * D_ * 2);
  __bf16* wops1_bf = (__bf16*)alloc((size_t)OPS_ * D_ * 2);
  __bf16* wnums_bf = (__bf16*)alloc((size_t)N_ * NUM_ * D_ * 2);
  __bf16* dec_bf   = (__bf16*)alloc((size_t)N_ * L_ * D_ * 2);
  __bf16* qscs_bf  = (__bf16*)alloc((size_t)N_ * L_ * 2 * D_ * 2); // cat(qs,cs)
  __bf16* wsoft_bf = (__bf16*)alloc((size_t)N_ * L_ * LP_ * 2);
  __bf16* feat_bf  = (__bf16*)alloc((size_t)N_ * L_ * D_ * 2);

  float* logits_out = (float*)d_out;
  float* hn_out     = (float*)d_out + LOGITS_ELEMS;

  auto cvt = [&](const float* s, __bf16* d, int n) {
    cvt_bf16_kernel<<<(n + 255) / 256, 256, 0, stream>>>(s, d, n);
  };

  // 1) fp32 -> bf16 weight / activation conversions
  cvt(Wih,   wih_bf,   G3_ * D_);
  cvt(Whh,   whh_bf,   G3_ * D_);
  cvt(attnW, attnw_bf, D_ * 2 * D_);
  cvt(fcW,   fcw_bf,   D_ * 2 * D_);
  cvt(enc,   enc_bf,   N_ * LP_ * D_);
  cvt(Wops1, wops1_bf, OPS_ * D_);
  cvt(Wnums, wnums_bf, N_ * NUM_ * D_);

  // 2) embedding gather
  embed_kernel<<<N_ * L_, 256, 0, stream>>>(ids, Wops0, Wnums, dec_bf);

  // 3) gi = dec_in @ Wih^T + bih   (M=1024, Nout=1536, K=512)
  gemm_awt_kernel<0><<<dim3(8, 24), 256, 0, stream>>>(
      dec_bf, D_, wih_bf, D_, bih, gi, G3_, nullptr, 0, D_);

  // 4) GRU scan (single WGP-resident workgroup); writes qs (bf16) and hn
  gru_kernel<<<1, 1024, 0, stream>>>(gi, h0, whh_bf, bhh, qscs_bf, hn_out);

  // 5) sx = qs @ attn_W[:, :D]^T   (M=1024, Nout=512, K=512)
  gemm_awt_kernel<0><<<dim3(8, 8), 256, 0, stream>>>(
      qscs_bf, 2 * D_, attnw_bf, 2 * D_, nullptr, sx, D_, nullptr, 0, D_);

  // 6) sm = enc @ attn_W[:, D:]^T + attn_Wb   (M=4096, Nout=512, K=512)
  gemm_awt_kernel<0><<<dim3(32, 8), 256, 0, stream>>>(
      enc_bf, D_, attnw_bf + D_, 2 * D_, attnWb, sm, D_, nullptr, 0, D_);

  // 7) additive-attention scores (async-LDS staged, hw tanh)
  scores_kernel<<<N_ * (L_ / QTILE_), 256, 0, stream>>>(
      sx, sm, attnV, attnVb, amask, scores);

  // 8) softmax over L' -> bf16 probabilities
  softmax_kernel<<<N_ * L_, 256, 0, stream>>>(scores, wsoft_bf);

  // 9) cs = w @ enc (batched), into cat buffer cols [512,1024)
  cs_kernel<<<dim3(8, 16), 256, 0, stream>>>(wsoft_bf, enc_bf, qscs_bf);

  // 10) feat = tanh(cat(qs,cs) @ fc_W^T + fc_b)  (M=1024, Nout=512, K=1024)
  gemm_awt_kernel<1><<<dim3(8, 8), 256, 0, stream>>>(
      qscs_bf, 2 * D_, fcw_bf, 2 * D_, fcb, nullptr, 0, feat_bf, D_, 2 * D_);

  // 11) logits = feat @ wn^T (batched, masked) -> d_out
  logits_kernel<<<N_, 256, 0, stream>>>(feat_bf, wops1_bf, wnums_bf, omask,
                                        logits_out);
}
